// ResidualLinearMerge_32530082300330
// MI455X (gfx1250) — matrix operand order repaired and checked by reading, not compiled, not run on hardware
//
#include <hip/hip_runtime.h>

// MI455X / gfx1250: wave32, WMMA 16x16x32 bf16 -> f32 accumulate.
// Log-depth scan via matrix powers; async-to-LDS double-buffered GEMM pipeline.

typedef unsigned short ushort_t;
typedef __attribute__((ext_vector_type(16))) __bf16 v16bf;
typedef __attribute__((ext_vector_type(8)))  float  v8f;

#define D_DIM 1024
#define KB    64            // K-panel per LDS tile
#define PITCH (KB + 8)      // 72 halves = 144B row pitch (16B aligned, de-conflicted)

struct __attribute__((aligned(16))) U128 { unsigned int w[4]; };
union FragCvt { U128 q[2]; v16bf v; };

__device__ __forceinline__ ushort_t f2bf(float f) {
    unsigned int u = __float_as_uint(f);
    unsigned int r = u + 0x7FFFu + ((u >> 16) & 1u);   // round-to-nearest-even
    return (ushort_t)(r >> 16);
}

// Async DMA: global (SGPR base + per-lane byte offset) -> LDS, 16B per lane.
// Tracked by ASYNCcnt. Low 32 bits of a generic LDS pointer = LDS byte address.
__device__ __forceinline__ void async_ld_b128(const void* sbase, unsigned gOffBytes,
                                              void* ldsPtr) {
    unsigned lds = (unsigned)(size_t)ldsPtr;
    asm volatile("global_load_async_to_lds_b128 %0, %1, %2"
                 :: "v"(lds), "v"(gOffBytes), "s"(sbase)
                 : "memory");
}
__device__ __forceinline__ void wait_async0() {
    asm volatile("s_wait_asynccnt 0" ::: "memory");
}

// A-fragment (16-bit, 16x32): lane m=L%16; halves 0..7 <- K=kb..kb+7, 8..15 <- K=kb+16..kb+23
__device__ __forceinline__ v16bf ld_frag_a(const ushort_t* base, int row0, int ks, int lane) {
    const ushort_t* r = base + (row0 + (lane & 15)) * PITCH + ks + ((lane & 16) ? 8 : 0);
    FragCvt f;
    f.q[0] = *reinterpret_cast<const U128*>(r);
    f.q[1] = *reinterpret_cast<const U128*>(r + 16);
    return f.v;
}
// B-fragment (16-bit, 32x16): lane n=L%16; halves 0..7 <- K=kb..kb+7, 8..15 <- K=kb+16..kb+23, kb=(L<16?0:8)
__device__ __forceinline__ v16bf ld_frag_b(const ushort_t* base, int row0, int ks, int lane) {
    const ushort_t* r = base + (row0 + (lane & 15)) * PITCH + ks + ((lane & 16) ? 8 : 0);
    FragCvt f;
    f.q[0] = *reinterpret_cast<const U128*>(r);
    f.q[1] = *reinterpret_cast<const U128*>(r + 16);
    return f.v;
}

// ---------------------------------------------------------------------------
// C[BM,1024-tile] = A @ B^T  (bf16 in, f32 accum, bf16 out)
// MT = WMMA m-tiles per wave. Block = 8 waves as 2(m) x 4(n):
//   block tile (MT*32) x 128, wave tile (MT*16) x 32.
// MT=2 -> 64x128 tile (128 WGs per 1024^2: serial power chain, max utilization)
// MT=4 -> 128x128 tile (batched / throughput GEMMs)
// Async-to-LDS double buffering, one barrier per K-panel.
// ---------------------------------------------------------------------------
template <int MT>
__global__ __launch_bounds__(256)
void k_gemm_nt(ushort_t* __restrict__ C, const ushort_t* __restrict__ A,
               const ushort_t* __restrict__ Bm, long strideA, long strideC) {
    constexpr int BM = MT * 32;
    __shared__ __align__(16) ushort_t As[2][BM * PITCH];
    __shared__ __align__(16) ushort_t Bs[2][128 * PITCH];

    const int tid  = threadIdx.x;
    const int lane = tid & 31;
    const int wave = tid >> 5;
    const int wm   = wave >> 2;   // 0..1
    const int wn   = wave & 3;    // 0..3
    const int rowBase = blockIdx.y * BM;
    const int colBase = blockIdx.x * 128;
    A += (long)blockIdx.z * strideA;
    C += (long)blockIdx.z * strideC;

    v8f acc[MT][2];
    const v8f vzero = {0.f, 0.f, 0.f, 0.f, 0.f, 0.f, 0.f, 0.f};
#pragma unroll
    for (int i = 0; i < MT; ++i)
#pragma unroll
        for (int j = 0; j < 2; ++j) acc[i][j] = vzero;

    auto issue = [&](int kk, int buf) {
#pragma unroll
        for (int t = 0; t < MT; ++t) {        // A panel: BM rows x 8 x (8 halves)
            int idx = tid + t * 256;
            int row = idx >> 3;
            int c8  = (idx & 7) << 3;
            async_ld_b128(A, (unsigned)(((rowBase + row) << 10) + kk + c8) * 2u,
                          &As[buf][row * PITCH + c8]);
        }
#pragma unroll
        for (int t = 0; t < 4; ++t) {         // B panel: 128 rows x 8 x (8 halves)
            int idx = tid + t * 256;
            int row = idx >> 3;
            int c8  = (idx & 7) << 3;
            async_ld_b128(Bm, (unsigned)(((colBase + row) << 10) + kk + c8) * 2u,
                          &Bs[buf][row * PITCH + c8]);
        }
    };

    issue(0, 0);
    for (int kk = 0; kk < D_DIM; kk += KB) {
        const int buf = (kk >> 6) & 1;
        wait_async0();              // my DMA writes into As/Bs[buf] have landed
        __syncthreads();            // everyone's landed; everyone done reading buf^1
        if (kk + KB < D_DIM) issue(kk + KB, buf ^ 1);
#pragma unroll
        for (int ks = 0; ks < KB; ks += 32) {
            v16bf af[MT], bfr[2];
#pragma unroll
            for (int i = 0; i < MT; ++i)
                af[i] = ld_frag_a(As[buf], wm * (MT * 16) + i * 16, ks, lane);
#pragma unroll
            for (int j = 0; j < 2; ++j) bfr[j] = ld_frag_b(Bs[buf], wn * 32 + j * 16, ks, lane);
#pragma unroll
            for (int i = 0; i < MT; ++i)
#pragma unroll
                for (int j = 0; j < 2; ++j)
                    acc[i][j] = __builtin_amdgcn_wmma_f32_16x16x32_bf16(
                        false, af[i], false, bfr[j], (short)0, acc[i][j], false, false);
        }
    }

    const int rOff = (lane & 16) ? 8 : 0;
    const int cOff = lane & 15;
#pragma unroll
    for (int i = 0; i < MT; ++i)
#pragma unroll
        for (int j = 0; j < 2; ++j)
#pragma unroll
            for (int r = 0; r < 8; ++r) {
                int row = rowBase + wm * (MT * 16) + i * 16 + r + rOff;
                int col = colBase + wn * 32 + j * 16 + cOff;
                C[(size_t)row * D_DIM + col] = f2bf(acc[i][j][r]);
            }
}

// ---------------------------------------------------------------------------
// Out_c[32,1024] = sum_{j=0..7} In_slice(8c+j)[32,1024] @ Mats[7-j][1024,1024]^T
// In element (slice s, row m, k) at In + s*sliceStride + m*rowStride + k.
// Block = one 32x128 output tile; 8 waves; flattened (j,kk) panel pipeline,
// Mat panels via async DMA, In panels async (bf16) or VALU-convert (f32).
// ---------------------------------------------------------------------------
template <typename InT, int OUT_F32>
__global__ __launch_bounds__(256)
void k_apply_reduce(void* __restrict__ OutPtr, const InT* __restrict__ In,
                    const ushort_t* __restrict__ Mats, long sliceStride, long rowStride) {
    __shared__ __align__(16) ushort_t InS[2][32 * PITCH];
    __shared__ __align__(16) ushort_t Ms[2][128 * PITCH];

    const int tid  = threadIdx.x;
    const int lane = tid & 31;
    const int wave = tid >> 5;            // column sub-tile 0..7
    const int colBase = blockIdx.x * 128;
    const int c = blockIdx.y;
    const int NPANEL = 8 * (D_DIM / KB);  // 8 j-slices x 16 K-panels

    v8f acc[2];
    const v8f vzero = {0.f, 0.f, 0.f, 0.f, 0.f, 0.f, 0.f, 0.f};
    acc[0] = vzero; acc[1] = vzero;

    auto stage = [&](int p, int buf) {
        const int j  = p >> 4;
        const int kk = (p & 15) * KB;
        const ushort_t* M = Mats + (size_t)(7 - j) * (size_t)D_DIM * D_DIM;
        const InT* S = In + (size_t)(c * 8 + j) * sliceStride;
#pragma unroll
        for (int t = 0; t < 4; ++t) {         // Mat panel 128 x KB, async DMA
            int idx = tid + t * 256;
            int row = idx >> 3;
            int c8  = (idx & 7) << 3;
            async_ld_b128(M, (unsigned)(((colBase + row) << 10) + kk + c8) * 2u,
                          &Ms[buf][row * PITCH + c8]);
        }
        if constexpr (sizeof(InT) == 4) {      // f32 -> bf16 convert via VALU
#pragma unroll
            for (int t = 0; t < 2; ++t) {
                int idx = tid + t * 256;       // 512: 32 rows x 16 float4
                int row = idx >> 4;
                int c4  = (idx & 15) << 2;
                float4 v = *reinterpret_cast<const float4*>(
                    &S[(size_t)row * rowStride + kk + c4]);
                ushort_t* dp = &InS[buf][row * PITCH + c4];
                dp[0] = f2bf(v.x); dp[1] = f2bf(v.y);
                dp[2] = f2bf(v.z); dp[3] = f2bf(v.w);
            }
        } else {                               // bf16: async DMA, 32 rows x 8 chunks
            int row = tid >> 3;
            int c8  = (tid & 7) << 3;
            async_ld_b128(S, (unsigned)((unsigned)row * (unsigned)rowStride + kk + c8) * 2u,
                          &InS[buf][row * PITCH + c8]);
        }
    };

    stage(0, 0);
    for (int p = 0; p < NPANEL; ++p) {
        const int buf = p & 1;
        wait_async0();
        __syncthreads();                       // also orders the VALU ds_stores (dscnt)
        if (p + 1 < NPANEL) stage(p + 1, buf ^ 1);
#pragma unroll
        for (int ks = 0; ks < KB; ks += 32) {
            v16bf b = ld_frag_b(Ms[buf], wave * 16, ks, lane);
#pragma unroll
            for (int i = 0; i < 2; ++i) {
                v16bf a = ld_frag_a(InS[buf], i * 16, ks, lane);
                acc[i] = __builtin_amdgcn_wmma_f32_16x16x32_bf16(
                    false, a, false, b, (short)0, acc[i], false, false);
            }
        }
    }

    const int rOff = (lane & 16) ? 8 : 0;
    const int cOff = lane & 15;
#pragma unroll
    for (int i = 0; i < 2; ++i)
#pragma unroll
        for (int r = 0; r < 8; ++r) {
            int row = i * 16 + r + rOff;
            int col = colBase + wave * 16 + cOff;
            size_t o = (size_t)c * 32 * D_DIM + (size_t)row * D_DIM + col;
            if constexpr (OUT_F32) ((float*)OutPtr)[o] = acc[i][r];
            else                   ((ushort_t*)OutPtr)[o] = f2bf(acc[i][r]);
        }
}

// ---------------------------------------------------------------------------
// Small helpers
// ---------------------------------------------------------------------------
__global__ void k_identity_bf16(ushort_t* dst) {
    int idx = blockIdx.x * 256 + threadIdx.x;
    dst[idx] = ((idx >> 10) == (idx & 1023)) ? (ushort_t)0x3F80 : (ushort_t)0;
}
__global__ void k_cast_bf16(ushort_t* dst, const float* src) {
    int idx = blockIdx.x * 256 + threadIdx.x;
    dst[idx] = f2bf(src[idx]);
}
__global__ void k_transpose_f32_bf16(ushort_t* dst, const float* src) {
    __shared__ float tile[32][33];
    int tx = threadIdx.x, ty = threadIdx.y;
    int x = blockIdx.x * 32 + tx, y0 = blockIdx.y * 32;
    for (int k = 0; k < 32; k += 8) tile[ty + k][tx] = src[(size_t)(y0 + ty + k) * D_DIM + x];
    __syncthreads();
    int x2 = blockIdx.y * 32 + tx, y2 = blockIdx.x * 32;
    for (int k = 0; k < 32; k += 8) dst[(size_t)(y2 + ty + k) * D_DIM + x2] = f2bf(tile[tx][ty + k]);
}
__global__ void k_transpose_bf16(ushort_t* dst, const ushort_t* src) {
    __shared__ ushort_t tile[32][33];
    int tx = threadIdx.x, ty = threadIdx.y;
    int x = blockIdx.x * 32 + tx, y0 = blockIdx.y * 32;
    for (int k = 0; k < 32; k += 8) tile[ty + k][tx] = src[(size_t)(y0 + ty + k) * D_DIM + x];
    __syncthreads();
    int x2 = blockIdx.y * 32 + tx, y2 = blockIdx.x * 32;
    for (int k = 0; k < 32; k += 8) dst[(size_t)(y2 + ty + k) * D_DIM + x2] = tile[tx][ty + k];
}

// ---------------------------------------------------------------------------
// Orchestration: h_final = sum_i x_i @ W1^T @ (W0^T)^{511-i}, radix-8, 3 levels.
//   Q_k = W0^k, P_k = W0^{8k}, S_k = W0^{64k};  G_j = Q_j @ W1
//   Y_c = sum_j x_{8c+j} @ G_{7-j}^T  ;  Z_e = sum_j Y_{8e+j} @ P_{7-j}^T
//   h   = sum_j Z_j @ S_{7-j}^T
// 20 serial 1024^3 GEMMs (vs 512 serial steps); ~85 MB bf16 scratch lives in L2.
// Serial-chain GEMMs use 64x128 tiles (128 WGs) to keep the chip busy on the
// critical path; the parallel batched GEMM uses 128x128 tiles.
// ---------------------------------------------------------------------------
extern "C" void kernel_launch(void* const* d_in, const int* in_sizes, int n_in,
                              void* d_out, int out_size, void* d_ws, size_t ws_size,
                              hipStream_t stream) {
    (void)in_sizes; (void)n_in; (void)out_size; (void)ws_size;
    const float* x  = (const float*)d_in[0];
    const float* W0 = (const float*)d_in[1];
    const float* W1 = (const float*)d_in[2];
    float* out = (float*)d_out;

    const size_t DD = (size_t)D_DIM * D_DIM;
    ushort_t* w   = (ushort_t*)d_ws;
    ushort_t* Q   = w;              // 9*DD
    ushort_t* P   = Q + 9 * DD;     // 9*DD
    ushort_t* Sm  = P + 9 * DD;     // 8*DD
    ushort_t* G   = Sm + 8 * DD;    // 8*DD
    ushort_t* R   = G + 8 * DD;     // W0^T
    ushort_t* W1T = R + DD;
    ushort_t* M1T = W1T + DD;       // (W0^8)^T
    ushort_t* M2T = M1T + DD;       // (W0^64)^T
    ushort_t* Y   = M2T + DD;       // 64*32*1024
    ushort_t* Z   = Y + (size_t)64 * 32 * D_DIM;  // 8*32*1024

    dim3 b256(256);
    dim3 gDD(DD / 256);
    dim3 tb(32, 8), tg(32, 32);
    dim3 ggSerial(8, 16, 1);        // MT=2: 64x128 tiles, 128 WGs
    dim3 ggBatchG(8, 8, 8);         // MT=4: 128x128 tiles, 512 WGs

    k_identity_bf16<<<gDD, b256, 0, stream>>>(Q);
    k_identity_bf16<<<gDD, b256, 0, stream>>>(P);
    k_identity_bf16<<<gDD, b256, 0, stream>>>(Sm);
    k_cast_bf16<<<gDD, b256, 0, stream>>>(Q + DD, W0);                 // Q_1 = W0
    k_transpose_f32_bf16<<<tg, tb, 0, stream>>>(R, W0);                // W0^T
    k_transpose_f32_bf16<<<tg, tb, 0, stream>>>(W1T, W1);              // W1^T

    for (int k = 1; k <= 7; ++k)                                       // Q_{k+1} = Q_k @ W0
        k_gemm_nt<2><<<ggSerial, b256, 0, stream>>>(
            Q + (size_t)(k + 1) * DD, Q + (size_t)k * DD, R, 0, 0);

    hipMemcpyAsync(P + DD, Q + 8 * DD, DD * sizeof(ushort_t), hipMemcpyDeviceToDevice, stream);
    k_transpose_bf16<<<tg, tb, 0, stream>>>(M1T, Q + 8 * DD);
    for (int k = 1; k <= 7; ++k)                                       // P_{k+1} = P_k @ W0^8
        k_gemm_nt<2><<<ggSerial, b256, 0, stream>>>(
            P + (size_t)(k + 1) * DD, P + (size_t)k * DD, M1T, 0, 0);

    hipMemcpyAsync(Sm + DD, P + 8 * DD, DD * sizeof(ushort_t), hipMemcpyDeviceToDevice, stream);
    k_transpose_bf16<<<tg, tb, 0, stream>>>(M2T, P + 8 * DD);
    for (int k = 1; k <= 6; ++k)                                       // S_{k+1} = S_k @ W0^64
        k_gemm_nt<2><<<ggSerial, b256, 0, stream>>>(
            Sm + (size_t)(k + 1) * DD, Sm + (size_t)k * DD, M2T, 0, 0);

    // G_j = Q_j @ W1 (batched over j, fully parallel)
    k_gemm_nt<4><<<ggBatchG, b256, 0, stream>>>(G, Q, W1T, (long)DD, (long)DD);

    // Level 1: 64 chunks over x (f32 in, bf16 out); x(b, i, d) -> slice i, row b
    k_apply_reduce<float, 0><<<dim3(8, 64), b256, 0, stream>>>(
        Y, x, G, (long)D_DIM, (long)(512 * D_DIM));
    // Level 2: 8 chunks over Y
    k_apply_reduce<ushort_t, 0><<<dim3(8, 8), b256, 0, stream>>>(
        Z, Y, P, (long)(32 * D_DIM), (long)D_DIM);
    // Level 3: final 32x1024 f32 output
    k_apply_reduce<ushort_t, 1><<<dim3(8, 1), b256, 0, stream>>>(
        out, Z, Sm, (long)(32 * D_DIM), (long)D_DIM);
}
